// Graphormer_10453950399030
// MI455X (gfx1250) — matrix operand order repaired and checked by reading, not compiled, not run on hardware
//
#include <hip/hip_runtime.h>
#include <hip/hip_bf16.h>
#include <math.h>
#include <stdint.h>

// ---------------------------------------------------------------------------
// Graphormer encoder layer for gfx1250 (MI455X).
// bf16 WMMA everywhere; GEMM tiles staged into LDS by the Tensor Data Mover
// (tensor_load_to_lds, TENSORcnt) with double buffering.
// ---------------------------------------------------------------------------

typedef __attribute__((ext_vector_type(16))) __bf16 v16bf;
typedef __attribute__((ext_vector_type(8)))  float  v8f;
typedef __attribute__((ext_vector_type(4)))  unsigned int u32x4;
typedef __attribute__((ext_vector_type(8)))  int   i32x8;
typedef __attribute__((ext_vector_type(4)))  int   i32x4;

#define FEAT     512
#define HEADS    8
#define HEAD_DIM 64
#define HIDDEN   2048
#define SEQ      512
#define BSZ      16
#define MTOT     (BSZ * SEQ)   // 8192 rows

#if __has_builtin(__builtin_amdgcn_tensor_load_to_lds) && \
    __has_builtin(__builtin_amdgcn_s_wait_tensorcnt)
#define USE_TDM 1
#else
#define USE_TDM 0
#endif

// ---------------------------------------------------------------------------
// TDM helper: async 2D tile copy (bf16) global -> LDS with row padding.
//   tile = tile_rows x 32 elements, element 2 B, row pitch in memory = K elems
//   LDS row stride = 32 + 8 halfs (pad 4 DWORDs every 16 DWORDs) = 80 B.
// ---------------------------------------------------------------------------
#if USE_TDM
__device__ __forceinline__ void tdm_load_tile_bf16(
    uint32_t lds_off, const __bf16* gptr, int rows_total, int K, int tile_rows) {
  const uint64_t ga = (uint64_t)(uintptr_t)gptr;
  u32x4 g0;
  g0[0] = 1u;                                   // count=1, user descriptor
  g0[1] = lds_off;                              // lds_addr (bytes)
  g0[2] = (uint32_t)(ga & 0xFFFFFFFFu);         // global_addr[31:0]
  g0[3] = (uint32_t)((ga >> 32) & 0x01FFFFFFu)  // global_addr[56:32]
          | (2u << 30);                         // type=2 ("image")
  i32x8 g1;
  g1[0] = (1 << 16)      // data_size = 1 (2-byte elements)
        | (1 << 20)      // pad_enable
        | (3 << 22)      // pad_interval: every 16 DWORDs (64 B)
        | (3 << 25);     // pad_amount: 4 DWORDs (16 B) -> row stride 80 B
  g1[1] = (int)((K & 0xFFFF) << 16);                         // tensor_dim0 lo
  g1[2] = (int)(((rows_total & 0xFFFF) << 16) |              // tensor_dim1 lo
                ((unsigned)K >> 16));                        // tensor_dim0 hi
  g1[3] = (int)((32 << 16) |                                 // tile_dim0 = 32
                (((unsigned)rows_total >> 16) & 0xFFFF));    // tensor_dim1 hi
  g1[4] = tile_rows;     // tile_dim1; tile_dim2 = 0 (unused)
  g1[5] = K;             // tensor_dim0_stride low 32 (elements)
  g1[6] = 0;
  g1[7] = 0;
  const i32x4 z4 = {0, 0, 0, 0};
#if defined(__clang_major__) && __clang_major__ >= 23
  const i32x8 z8 = {0, 0, 0, 0, 0, 0, 0, 0};
  __builtin_amdgcn_tensor_load_to_lds(g0, g1, z4, z4, z8, 0);
#else
  __builtin_amdgcn_tensor_load_to_lds(g0, g1, z4, z4, 0);
#endif
}
#endif

// ---------------------------------------------------------------------------
// Generic GEMM:  Y[M,N] = act( X[M,K](bf16) @ W[N,K](bf16)^T + bias (+resid) )
// BM=128, BN=128, BK=32; 256 threads = 8 waves (4x2), each wave 32x64 out
// via 2x4 v_wmma_f32_16x16x32_bf16. LDS tiles TDM-loaded, double buffered.
// ---------------------------------------------------------------------------
#define BM 128
#define BN 128
#define BK 32
#define LDSTR 40   // LDS row stride in halfs (80 B, 16B-aligned, pad 8)

template <typename TOUT, bool RELU>
__global__ __launch_bounds__(256) void gemm_bias_kernel(
    const __bf16* __restrict__ X, const __bf16* __restrict__ W,
    const float* __restrict__ bias, const float* __restrict__ resid,
    TOUT* __restrict__ Y, int M, int N, int K) {
  __shared__ alignas(16) __bf16 Asm[2][BM][LDSTR];
  __shared__ alignas(16) __bf16 Bsm[2][BN][LDSTR];

  const int tid  = threadIdx.x;
  const int lane = tid & 31;
  const int wave = tid >> 5;
  const int half = lane >> 4;
  const int l15  = lane & 15;

  const int m0 = blockIdx.x * BM;
  const int n0 = blockIdx.y * BN;
  const int wr = wave & 3;
  const int wc = wave >> 2;

  v8f acc[2][4];
#pragma unroll
  for (int t = 0; t < 2; ++t)
#pragma unroll
    for (int u = 0; u < 4; ++u)
#pragma unroll
      for (int r = 0; r < 8; ++r) acc[t][u][r] = 0.0f;

  const int nk = K / BK;

#if USE_TDM
  if (wave == 0) {
    tdm_load_tile_bf16((uint32_t)(uintptr_t)&Asm[0][0][0],
                       X + (size_t)m0 * K, M, K, BM);
    tdm_load_tile_bf16((uint32_t)(uintptr_t)&Bsm[0][0][0],
                       W + (size_t)n0 * K, N, K, BN);
  }
#endif

  for (int i = 0; i < nk; ++i) {
    const int buf = i & 1;
#if USE_TDM
    if (wave == 0) {
      if (i + 1 < nk) {
        const int k1 = (i + 1) * BK;
        tdm_load_tile_bf16((uint32_t)(uintptr_t)&Asm[buf ^ 1][0][0],
                           X + (size_t)m0 * K + k1, M, K, BM);
        tdm_load_tile_bf16((uint32_t)(uintptr_t)&Bsm[buf ^ 1][0][0],
                           W + (size_t)n0 * K + k1, N, K, BN);
        __builtin_amdgcn_s_wait_tensorcnt(2);   // current pair complete
      } else {
        __builtin_amdgcn_s_wait_tensorcnt(0);
      }
    }
#else
    {
      const int k0 = i * BK;
      const int r  = tid >> 1;
      const int cg = (tid & 1) * 16;
      const __bf16* sa = X + (size_t)(m0 + r) * K + k0 + cg;
#pragma unroll
      for (int j = 0; j < 16; ++j) Asm[buf][r][cg + j] = sa[j];
      const __bf16* sb = W + (size_t)(n0 + r) * K + k0 + cg;
#pragma unroll
      for (int j = 0; j < 16; ++j) Bsm[buf][r][cg + j] = sb[j];
    }
#endif
    __syncthreads();

    // ---- fragments: contiguous k = 16*half + c, same order in A and B ----
    v16bf afrag[2], bfrag[4];
#pragma unroll
    for (int t = 0; t < 2; ++t) {
      const int row = wr * 32 + t * 16 + l15;
#pragma unroll
      for (int c = 0; c < 16; ++c)
        afrag[t][c] = Asm[buf][row][half * 16 + c];
    }
#pragma unroll
    for (int u = 0; u < 4; ++u) {
      const int col = wc * 64 + u * 16 + l15;
#pragma unroll
      for (int c = 0; c < 16; ++c)
        bfrag[u][c] = Bsm[buf][col][half * 16 + c];
    }

#pragma unroll
    for (int t = 0; t < 2; ++t)
#pragma unroll
      for (int u = 0; u < 4; ++u)
        acc[t][u] = __builtin_amdgcn_wmma_f32_16x16x32_bf16(
            false, afrag[t], false, bfrag[u], (short)0, acc[t][u], false, false);
    __syncthreads();
  }

  // ---- epilogue: bias (+resid) (+relu) ----
#pragma unroll
  for (int t = 0; t < 2; ++t) {
#pragma unroll
    for (int u = 0; u < 4; ++u) {
      const int n = n0 + wc * 64 + u * 16 + l15;
      const float bn = bias[n];
#pragma unroll
      for (int r = 0; r < 8; ++r) {
        const int m = m0 + wr * 32 + t * 16 + 8 * half + r;
        float v = acc[t][u][r] + bn;
        if (resid) v += resid[(size_t)m * N + n];
        if (RELU) v = fmaxf(v, 0.0f);
        Y[(size_t)m * N + n] = (TOUT)v;
      }
    }
  }
}

// ---------------------------------------------------------------------------
// Flash-style biased/masked multi-head attention (unchanged from last round).
// ---------------------------------------------------------------------------
__global__ __launch_bounds__(256) void attn_kernel(
    const __bf16* __restrict__ q, const __bf16* __restrict__ k,
    const __bf16* __restrict__ v, const float* __restrict__ bias,
    const int* __restrict__ mask, __bf16* __restrict__ ctx) {
  const int b    = blockIdx.z;
  const int h    = blockIdx.y;
  const int tid  = threadIdx.x;
  const int lane = tid & 31;
  const int wave = tid >> 5;
  const int half = lane >> 4;
  const int l15  = lane & 15;
  const int qbase = blockIdx.x * 128 + wave * 16;

  __shared__ alignas(16) __bf16 Klds[32][72];    // [kv][d]
  __shared__ alignas(16) __bf16 Vtlds[64][40];   // [d][kv] (transposed)
  __shared__ alignas(16) __bf16 Plds[8][16][40]; // per-wave P scratch

  v16bf qa[2];
  const size_t qrow = ((size_t)b * SEQ + qbase + l15) * FEAT + h * HEAD_DIM;
#pragma unroll
  for (int ks = 0; ks < 2; ++ks)
#pragma unroll
    for (int c = 0; c < 16; ++c)
      qa[ks][c] = q[qrow + ks * 32 + half * 16 + c];

  v8f cacc[4];
#pragma unroll
  for (int t = 0; t < 4; ++t)
#pragma unroll
    for (int r = 0; r < 8; ++r) cacc[t][r] = 0.0f;
  float mrun[8], lrun[8];
#pragma unroll
  for (int r = 0; r < 8; ++r) { mrun[r] = -INFINITY; lrun[r] = 0.0f; }

  for (int kv0 = 0; kv0 < SEQ; kv0 += 32) {
    {
      const int r  = tid >> 3;
      const int cg = (tid & 7) * 8;
      const size_t off = ((size_t)b * SEQ + kv0 + r) * FEAT + h * HEAD_DIM + cg;
#pragma unroll
      for (int j = 0; j < 8; ++j) Klds[r][cg + j] = k[off + j];
#pragma unroll
      for (int j = 0; j < 8; ++j) Vtlds[cg + j][r] = v[off + j];
    }
    __syncthreads();

    v8f s[2];
#pragma unroll
    for (int u = 0; u < 2; ++u)
#pragma unroll
      for (int r = 0; r < 8; ++r) s[u][r] = 0.0f;
#pragma unroll
    for (int u = 0; u < 2; ++u) {
      const int kvr = u * 16 + l15;
#pragma unroll
      for (int ks = 0; ks < 2; ++ks) {
        v16bf bf;
#pragma unroll
        for (int c = 0; c < 16; ++c)
          bf[c] = Klds[kvr][ks * 32 + half * 16 + c];
        s[u] = __builtin_amdgcn_wmma_f32_16x16x32_bf16(
            false, qa[ks], false, bf, (short)0, s[u], false, false);
      }
    }

    float sm[2][8];
#pragma unroll
    for (int u = 0; u < 2; ++u) {
      const int kc = kv0 + u * 16 + l15;
#pragma unroll
      for (int r = 0; r < 8; ++r) {
        const int qr = qbase + 8 * half + r;
        const size_t rowoff = ((size_t)b * SEQ + qr) * SEQ + kc;
        float val = s[u][r] * 8.0f + bias[rowoff * HEADS + h];
        if (mask[rowoff] != 0) val = -INFINITY;
        sm[u][r] = val;
      }
    }

#pragma unroll
    for (int r = 0; r < 8; ++r) {
      float vmx = fmaxf(sm[0][r], sm[1][r]);
#pragma unroll
      for (int off = 8; off >= 1; off >>= 1)
        vmx = fmaxf(vmx, __shfl_xor(vmx, off, 32));
      const float mnew = fmaxf(mrun[r], vmx);
      const float alpha = (mnew == -INFINITY) ? 0.0f : __expf(mrun[r] - mnew);
      mrun[r] = mnew;
      lrun[r] *= alpha;
#pragma unroll
      for (int t = 0; t < 4; ++t) cacc[t][r] *= alpha;

      const float mm = (mnew == -INFINITY) ? 0.0f : mnew;
      const float p0 = __expf(sm[0][r] - mm);
      const float p1 = __expf(sm[1][r] - mm);
      Plds[wave][8 * half + r][l15]      = (__bf16)p0;
      Plds[wave][8 * half + r][16 + l15] = (__bf16)p1;
      float pr = p0 + p1;
#pragma unroll
      for (int off = 8; off >= 1; off >>= 1) pr += __shfl_xor(pr, off, 32);
      lrun[r] += pr;
    }

    asm volatile("s_wait_dscnt 0" ::: "memory");

    v16bf pa;
#pragma unroll
    for (int c = 0; c < 16; ++c)
      pa[c] = Plds[wave][l15][half * 16 + c];
#pragma unroll
    for (int t = 0; t < 4; ++t) {
      v16bf vb;
#pragma unroll
      for (int c = 0; c < 16; ++c)
        vb[c] = Vtlds[t * 16 + l15][half * 16 + c];
      cacc[t] = __builtin_amdgcn_wmma_f32_16x16x32_bf16(
          false, pa, false, vb, (short)0, cacc[t], false, false);
    }
    __syncthreads();
  }

#pragma unroll
  for (int r = 0; r < 8; ++r) {
    const float inv = 1.0f / lrun[r];
    const int qr = qbase + 8 * half + r;
#pragma unroll
    for (int t = 0; t < 4; ++t) {
      const int d = t * 16 + l15;
      ctx[((size_t)b * SEQ + qr) * FEAT + h * HEAD_DIM + d] =
          (__bf16)(cacc[t][r] * inv);
    }
  }
}

// ---------------------------------------------------------------------------
// LayerNorm (wave per row); optionally also writes a bf16 copy.
// ---------------------------------------------------------------------------
__global__ __launch_bounds__(256) void layernorm_kernel(
    const float* __restrict__ X, const float* __restrict__ g,
    const float* __restrict__ bta, float* __restrict__ Y,
    __bf16* __restrict__ Ybf, int ncols) {
  const int row  = blockIdx.x * 8 + (threadIdx.x >> 5);
  const int lane = threadIdx.x & 31;
  const float* x = X + (size_t)row * ncols;
  float s = 0.0f, s2 = 0.0f;
  for (int j = lane; j < ncols; j += 32) {
    const float v = x[j];
    s += v; s2 += v * v;
  }
#pragma unroll
  for (int off = 16; off >= 1; off >>= 1) {
    s  += __shfl_xor(s, off, 32);
    s2 += __shfl_xor(s2, off, 32);
  }
  const float mean = s / ncols;
  const float var  = s2 / ncols - mean * mean;
  const float rstd = rsqrtf(var + 1e-5f);
  float* y = Y + (size_t)row * ncols;
  for (int j = lane; j < ncols; j += 32) {
    const float o = (x[j] - mean) * rstd * g[j] + bta[j];
    y[j] = o;
    if (Ybf) Ybf[(size_t)row * ncols + j] = (__bf16)o;
  }
}

// ---------------------------------------------------------------------------
// fp32 -> bf16 convert (weights / activations fed to TDM-staged GEMMs)
// ---------------------------------------------------------------------------
__global__ __launch_bounds__(256) void f32_to_bf16_kernel(
    const float* __restrict__ x, __bf16* __restrict__ y, int n) {
  const int i = blockIdx.x * 256 + threadIdx.x;
  if (i < n) y[i] = (__bf16)x[i];
}

// ---------------------------------------------------------------------------
// Host launcher
// ---------------------------------------------------------------------------
extern "C" void kernel_launch(void* const* d_in, const int* in_sizes, int n_in,
                              void* d_out, int out_size, void* d_ws,
                              size_t ws_size, hipStream_t stream) {
  const float* nfeat = (const float*)d_in[0];
  const float* abias = (const float*)d_in[1];
  const int*   amask = (const int*)d_in[2];
  const float* Wq = (const float*)d_in[3];  const float* bq = (const float*)d_in[4];
  const float* Wk = (const float*)d_in[5];  const float* bk = (const float*)d_in[6];
  const float* Wv = (const float*)d_in[7];  const float* bv = (const float*)d_in[8];
  const float* Wo = (const float*)d_in[9];  const float* bo = (const float*)d_in[10];
  const float* W1 = (const float*)d_in[11]; const float* b1 = (const float*)d_in[12];
  const float* W2 = (const float*)d_in[13]; const float* b2 = (const float*)d_in[14];
  const float* ln1_g = (const float*)d_in[15]; const float* ln1_b = (const float*)d_in[16];
  const float* ln2_g = (const float*)d_in[17]; const float* ln2_b = (const float*)d_in[18];

  // ---- workspace carve-up (bytes) ----
  char* ws = (char*)d_ws;
  const size_t BF_FEAT = (size_t)MTOT * FEAT * sizeof(__bf16);    // 8 MB
  const size_t F_FEAT  = (size_t)MTOT * FEAT * sizeof(float);     // 16 MB
  const size_t BF_HID  = (size_t)MTOT * HIDDEN * sizeof(__bf16);  // 32 MB
  const size_t BF_W    = (size_t)FEAT * FEAT * sizeof(__bf16);    // 0.5 MB
  const size_t BF_W1   = (size_t)HIDDEN * FEAT * sizeof(__bf16);  // 2 MB

  size_t off = 0;
  __bf16* q_bf    = (__bf16*)(ws + off); off += BF_FEAT;
  __bf16* k_bf    = (__bf16*)(ws + off); off += BF_FEAT;
  __bf16* v_bf    = (__bf16*)(ws + off); off += BF_FEAT;
  __bf16* ctx_bf  = (__bf16*)(ws + off); off += BF_FEAT;
  __bf16* x_bf    = (__bf16*)(ws + off); off += BF_FEAT;  // nfeat, bf16
  __bf16* x1_bf   = (__bf16*)(ws + off); off += BF_FEAT;  // ln1 out, bf16
  float*  y1      = (float*)(ws + off);  off += F_FEAT;
  float*  x1      = (float*)(ws + off);  off += F_FEAT;
  float*  y2      = (float*)(ws + off);  off += F_FEAT;
  __bf16* h_bf    = (__bf16*)(ws + off); off += BF_HID;
  __bf16* Wq_bf   = (__bf16*)(ws + off); off += BF_W;
  __bf16* Wk_bf   = (__bf16*)(ws + off); off += BF_W;
  __bf16* Wv_bf   = (__bf16*)(ws + off); off += BF_W;
  __bf16* Wo_bf   = (__bf16*)(ws + off); off += BF_W;
  __bf16* W1_bf   = (__bf16*)(ws + off); off += BF_W1;
  __bf16* W2_bf   = (__bf16*)(ws + off); off += BF_W1;

  const dim3 blk(256);
  const dim3 gFeat(MTOT / BM, FEAT / BN);   // 64 x 4
  const dim3 gHid(MTOT / BM, HIDDEN / BN);  // 64 x 16
  const dim3 gAttn(SEQ / 128, HEADS, BSZ);  // 4 x 8 x 16
  const dim3 gLN(MTOT / 8);                 // 1024

  // ---- one-time-per-call bf16 conversions (tiny vs. 280 MB total stream) --
  f32_to_bf16_kernel<<<(MTOT * FEAT) / 256, blk, 0, stream>>>(nfeat, x_bf, MTOT * FEAT);
  f32_to_bf16_kernel<<<(FEAT * FEAT) / 256, blk, 0, stream>>>(Wq, Wq_bf, FEAT * FEAT);
  f32_to_bf16_kernel<<<(FEAT * FEAT) / 256, blk, 0, stream>>>(Wk, Wk_bf, FEAT * FEAT);
  f32_to_bf16_kernel<<<(FEAT * FEAT) / 256, blk, 0, stream>>>(Wv, Wv_bf, FEAT * FEAT);
  f32_to_bf16_kernel<<<(FEAT * FEAT) / 256, blk, 0, stream>>>(Wo, Wo_bf, FEAT * FEAT);
  f32_to_bf16_kernel<<<(HIDDEN * FEAT) / 256, blk, 0, stream>>>(W1, W1_bf, HIDDEN * FEAT);
  f32_to_bf16_kernel<<<(FEAT * HIDDEN) / 256, blk, 0, stream>>>(W2, W2_bf, FEAT * HIDDEN);

  // ---- QKV projections ----
  gemm_bias_kernel<__bf16, false><<<gFeat, blk, 0, stream>>>(
      x_bf, Wq_bf, bq, nullptr, q_bf, MTOT, FEAT, FEAT);
  gemm_bias_kernel<__bf16, false><<<gFeat, blk, 0, stream>>>(
      x_bf, Wk_bf, bk, nullptr, k_bf, MTOT, FEAT, FEAT);
  gemm_bias_kernel<__bf16, false><<<gFeat, blk, 0, stream>>>(
      x_bf, Wv_bf, bv, nullptr, v_bf, MTOT, FEAT, FEAT);

  // ---- biased masked attention ----
  attn_kernel<<<gAttn, blk, 0, stream>>>(q_bf, k_bf, v_bf, abias, amask, ctx_bf);

  // ---- output projection + residual, LN1 (fp32 + bf16 outputs) ----
  gemm_bias_kernel<float, false><<<gFeat, blk, 0, stream>>>(
      ctx_bf, Wo_bf, bo, nfeat, y1, MTOT, FEAT, FEAT);
  layernorm_kernel<<<gLN, blk, 0, stream>>>(y1, ln1_g, ln1_b, x1, x1_bf, FEAT);

  // ---- FFN ----
  gemm_bias_kernel<__bf16, true><<<gHid, blk, 0, stream>>>(
      x1_bf, W1_bf, b1, nullptr, h_bf, MTOT, HIDDEN, FEAT);
  gemm_bias_kernel<float, false><<<gFeat, blk, 0, stream>>>(
      h_bf, W2_bf, b2, x1, y2, MTOT, FEAT, HIDDEN);
  layernorm_kernel<<<gLN, blk, 0, stream>>>(y2, ln2_g, ln2_b, (float*)d_out,
                                            nullptr, FEAT);
}
